// Decoder_39591008535099
// MI455X (gfx1250) — matrix operand order repaired and checked by reading, not compiled, not run on hardware
//
#include <hip/hip_runtime.h>
#include <hip/hip_bf16.h>
#include <math.h>

typedef __bf16 bf16;
typedef __attribute__((ext_vector_type(16))) __bf16 v16bf;
typedef __attribute__((ext_vector_type(8)))  float  v8f;

#define EPSV 1e-5f
#define T_STEPS 12
#define BTOT 4096
#define GRP 256
#define HD 128

// ---------------------------------------------------------------------------
// bf16 WMMA GEMM, one wave per 16x32 output tile (2 N-tiles, A reuse).
// A: row-major bf16 activations (lda); fragment layout per CDNA5 16-bit A:
//    lanes 0-15 carry K 0-7/16-23, lanes 16-31 carry K 8-15/24-31 — per lane
//    this is two contiguous 16B chunks, so loads coalesce into 2x b128.
// B: pre-swizzled fragments ((tile_n*Ksteps + ks)*32 + lane)*16 -> 2x b128.
// C: fp32 (+ optional per-column bias). blockIdx.y = group (strided A/C).
// ---------------------------------------------------------------------------
__global__ __launch_bounds__(256) void k_gemm(
    const bf16* __restrict__ A, int lda, long long Astride,
    const bf16* __restrict__ Bf,
    const float* __restrict__ bias,
    float* __restrict__ C, int ldc, long long Cstride,
    int Ntiles, int Ksteps)
{
  int g      = blockIdx.y;
  int w      = blockIdx.x * 8 + (threadIdx.x >> 5);  // wave id (uniform)
  int lane   = threadIdx.x & 31;
  int npairs = Ntiles >> 1;
  int tile_m  = w / npairs;
  int tile_n0 = (w % npairs) * 2;
  int m    = lane & 15;
  int half = lane >> 4;

  const bf16* arow  = A + g * Astride + (long long)(tile_m * 16 + m) * lda;
  const bf16* bptr0 = Bf + ((long long)(tile_n0 * Ksteps) * 32 + lane) * 16;
  const bf16* bptr1 = Bf + ((long long)((tile_n0 + 1) * Ksteps) * 32 + lane) * 16;

  int kp[8];
#pragma unroll
  for (int p = 0; p < 8; ++p)
    kp[p] = ((p & 4) << 2) + half * 8 + ((p & 3) << 1);  // k of bf16 pair p

  v8f acc0 = {};
  v8f acc1 = {};
  for (int ks = 0; ks < Ksteps; ++ks) {
    union { v16bf v; unsigned u[8]; } ua;
    int base = ks * 32;
#pragma unroll
    for (int p = 0; p < 8; ++p)
      ua.u[p] = *reinterpret_cast<const unsigned*>(arow + base + kp[p]);
    v16bf bv0 = *reinterpret_cast<const v16bf*>(bptr0);
    v16bf bv1 = *reinterpret_cast<const v16bf*>(bptr1);
    bptr0 += 32 * 16;
    bptr1 += 32 * 16;
    acc0 = __builtin_amdgcn_wmma_f32_16x16x32_bf16(
        false, ua.v, false, bv0, (short)0, acc0, false, false);
    acc1 = __builtin_amdgcn_wmma_f32_16x16x32_bf16(
        false, ua.v, false, bv1, (short)0, acc1, false, false);
  }

  int n0 = tile_n0 * 16 + m;
  float bb0 = bias ? bias[n0] : 0.0f;
  float bb1 = bias ? bias[n0 + 16] : 0.0f;
  float* crow = C + g * Cstride + (long long)(tile_m * 16 + half * 8) * ldc + n0;
#pragma unroll
  for (int r = 0; r < 8; ++r) {
    crow[(long long)r * ldc]      = acc0[r] + bb0;
    crow[(long long)r * ldc + 16] = acc1[r] + bb1;
  }
}

// Pre-swizzle fp32 weights (optionally concatenated [W0;W1] along K) into the
// bf16 B-fragment layout consumed by k_gemm.
__global__ void k_swizzle(const float* __restrict__ W0,
                          const float* __restrict__ W1, int ksplit, int ldw,
                          int Ksteps, bf16* __restrict__ out, int total)
{
  int idx = blockIdx.x * blockDim.x + threadIdx.x;
  if (idx >= total) return;
  int e      = idx & 15;
  int lane   = (idx >> 4) & 31;
  int t      = idx >> 9;
  int ks     = t % Ksteps;
  int tile_n = t / Ksteps;
  int n = tile_n * 16 + (lane & 15);
  int k = ks * 32 + ((e >> 3) << 4) + (lane >> 4) * 8 + (e & 7);
  float v = (k < ksplit) ? W0[(long long)k * ldw + n]
                         : W1[(long long)(k - ksplit) * ldw + n];
  out[idx] = (bf16)v;
}

// Fold Wp_se through Wp1[:64] (emb is affine in rel):  Wc (2x512), cbias(512)
__global__ void k_wc(const float* __restrict__ Wpse, const float* __restrict__ bpse,
                     const float* __restrict__ Wp1, const float* __restrict__ bp1,
                     float* __restrict__ Wc, float* __restrict__ cbias)
{
  int n = blockIdx.x * blockDim.x + threadIdx.x;  // 512
  float c0 = 0.f, c1 = 0.f, cb = 0.f;
  for (int j = 0; j < 64; ++j) {
    float w = Wp1[(long long)j * 512 + n];
    c0 += Wpse[j] * w;
    c1 += Wpse[64 + j] * w;
    cb += bpse[j] * w;
  }
  Wc[n] = c0; Wc[512 + n] = c1; cbias[n] = cb + bp1[n];
}

__global__ void k_biascomb(const float* a, const float* b, float* o)
{
  int i = blockIdx.x * blockDim.x + threadIdx.x;
  o[i] = a[i] + b[i];
}

__global__ void k_zero(float* p, int n)
{
  int i = blockIdx.x * blockDim.x + threadIdx.x;
  if (i < n) p[i] = 0.f;
}

__global__ void k_copy(const float* __restrict__ s, float* __restrict__ d, int n)
{
  int i = blockIdx.x * blockDim.x + threadIdx.x;
  if (i < n) d[i] = s[i];
}

// t=0 state init: c, last_pos, A = [bf16(dec_in) | bf16(h0)]
__global__ __launch_bounds__(128) void k_init(
    const float* __restrict__ h0, const float* __restrict__ c0,
    const float* __restrict__ lp, const float* __restrict__ lpr,
    const float* __restrict__ Wse, const float* __restrict__ bse,
    float* __restrict__ c, bf16* __restrict__ acat, float* __restrict__ lastpos)
{
  int b = blockIdx.x, j = threadIdx.x;
  c[b * HD + j] = c0[b * HD + j];
  acat[b * 192 + 64 + j] = (bf16)h0[b * HD + j];
  if (j < 64) {
    float d = lpr[b * 2] * Wse[j] + lpr[b * 2 + 1] * Wse[64 + j] + bse[j];
    acat[b * 192 + j] = (bf16)d;
  }
  if (j < 2) lastpos[b * 2 + j] = lp[b * 2 + j];
}

// LSTM cell + hidden2pos + position update + next spatial embedding (fused).
__global__ __launch_bounds__(128) void k_cell(
    const float* __restrict__ gates, float* __restrict__ c,
    bf16* __restrict__ hcat, bf16* __restrict__ acat,
    float* __restrict__ lastpos, float* __restrict__ traj,
    const float* __restrict__ Whp, const float* __restrict__ bhp,
    const float* __restrict__ Wse, const float* __restrict__ bse)
{
  __shared__ float s0[128], s1[128], rel[2];
  int b = blockIdx.x, j = threadIdx.x;
  const float* gr = gates + (long long)b * 512;
  float gi = gr[j], gf = gr[128 + j], gg = gr[256 + j], go = gr[384 + j];
  float cn = (1.f / (1.f + __expf(-gf))) * c[b * HD + j] +
             (1.f / (1.f + __expf(-gi))) * tanhf(gg);
  float hn = (1.f / (1.f + __expf(-go))) * tanhf(cn);
  c[b * HD + j] = cn;
  hcat[(long long)b * 256 + j] = (bf16)hn;
  s0[j] = hn * Whp[2 * j];
  s1[j] = hn * Whp[2 * j + 1];
  __syncthreads();
  for (int st = 64; st > 0; st >>= 1) {
    if (j < st) { s0[j] += s0[j + st]; s1[j] += s1[j + st]; }
    __syncthreads();
  }
  if (j == 0) {
    float r0 = s0[0] + bhp[0], r1 = s1[0] + bhp[1];
    rel[0] = r0; rel[1] = r1;
    traj[b * 2] = r0; traj[b * 2 + 1] = r1;
    lastpos[b * 2] += r0; lastpos[b * 2 + 1] += r1;
  }
  __syncthreads();
  if (j < 64) {
    float d = rel[0] * Wse[j] + rel[1] * Wse[64 + j] + bse[j];
    acat[b * 192 + j] = (bf16)d;
  }
}

// Pool layer 1 (post hproj): add folded rel terms, per-group BN(256 rows)+ReLU,
// emit bf16 x1. One block = (group, 256-column tile).
__global__ __launch_bounds__(256) void k_pool1(
    const float* __restrict__ hproj, const float* __restrict__ lastpos,
    const float* __restrict__ Wc, const float* __restrict__ gp1,
    const float* __restrict__ betap1, bf16* __restrict__ x1)
{
  __shared__ float pos[32];
  int g = blockIdx.y;
  int n = blockIdx.x * 256 + threadIdx.x;
  if (threadIdx.x < 32) pos[threadIdx.x] = lastpos[g * 32 + threadIdx.x];
  __syncthreads();
  float w0 = Wc[n], w1 = Wc[512 + n];
  float hp[16];
#pragma unroll
  for (int a = 0; a < 16; ++a)
    hp[a] = hproj[(long long)(g * 16 + a) * 512 + n];
  float sum = 0.f, sq = 0.f;
  for (int a = 0; a < 16; ++a)
    for (int b = 0; b < 16; ++b) {
      float rx = pos[b * 2] - pos[a * 2], ry = pos[b * 2 + 1] - pos[a * 2 + 1];
      float v = hp[a] + rx * w0 + ry * w1;
      sum += v; sq += v * v;
    }
  float mean  = sum * (1.f / 256.f);
  float var   = sq * (1.f / 256.f) - mean * mean;
  float scale = gp1[n] * rsqrtf(var + EPSV);
  float shift = betap1[n] - mean * scale;
  bf16* xr = x1 + (long long)g * 256 * 512 + n;
  for (int a = 0; a < 16; ++a)
    for (int b = 0; b < 16; ++b) {
      float rx = pos[b * 2] - pos[a * 2], ry = pos[b * 2 + 1] - pos[a * 2 + 1];
      float v = hp[a] + rx * w0 + ry * w1;
      xr[(long long)(a * 16 + b) * 512] = (bf16)fmaxf(v * scale + shift, 0.f);
    }
}

// Pool layer 2 epilogue: per-group BN + ReLU + max over b, into hcat[:,128:256]
__global__ __launch_bounds__(128) void k_pool_bnmax(
    const float* __restrict__ x2, const float* __restrict__ gp2,
    const float* __restrict__ betap2, bf16* __restrict__ hcat)
{
  int g = blockIdx.x, cidx = threadIdx.x;
  const float* xr = x2 + (long long)g * 256 * 128 + cidx;
  float sum = 0.f, sq = 0.f;
  for (int r = 0; r < 256; ++r) { float v = xr[(long long)r * 128]; sum += v; sq += v * v; }
  float mean  = sum * (1.f / 256.f);
  float var   = sq * (1.f / 256.f) - mean * mean;
  float scale = gp2[cidx] * rsqrtf(var + EPSV);
  float shift = betap2[cidx] - mean * scale;
  for (int a = 0; a < 16; ++a) {
    float m = 0.f;  // relu(x) >= 0
    for (int b = 0; b < 16; ++b)
      m = fmaxf(m, fmaxf(xr[(long long)(a * 16 + b) * 128] * scale + shift, 0.f));
    hcat[(long long)(g * 16 + a) * 256 + 128 + cidx] = (bf16)m;
  }
}

// Batch-wide column sums / sums-of-squares (merge BN), atomic partial reduce.
__global__ __launch_bounds__(256) void k_colstats(
    const float* __restrict__ X, int N, float* __restrict__ stats)
{
  __shared__ float ss[256], qq[256];
  int c     = blockIdx.x * 64 + (threadIdx.x & 63);
  int chunk = threadIdx.x >> 6;
  int r0    = blockIdx.y * 1024 + chunk * 256;
  float s = 0.f, q = 0.f;
  const float* xp = X + (long long)r0 * N + c;
  for (int r = 0; r < 256; ++r) { float v = xp[(long long)r * N]; s += v; q += v * v; }
  ss[threadIdx.x] = s; qq[threadIdx.x] = q;
  __syncthreads();
  if (chunk == 0) {
    int t = threadIdx.x;
    s = ss[t] + ss[t + 64] + ss[t + 128] + ss[t + 192];
    q = qq[t] + qq[t + 64] + qq[t + 128] + qq[t + 192];
    atomicAdd(&stats[c], s);
    atomicAdd(&stats[N + c], q);
  }
}

// Batch BN + ReLU; emit bf16 into the next GEMM A-buffer (and optional fp32).
__global__ void k_bnrelu(const float* __restrict__ X, int N,
                         const float* __restrict__ stats,
                         const float* __restrict__ gamma,
                         const float* __restrict__ beta,
                         bf16* __restrict__ obf, int ldo, int ooff,
                         float* __restrict__ of)
{
  long long idx = (long long)blockIdx.x * blockDim.x + threadIdx.x;
  int b = (int)(idx / N), n = (int)(idx % N);
  float mean = stats[n] * (1.f / BTOT);
  float var  = stats[N + n] * (1.f / BTOT) - mean * mean;
  float y = fmaxf((X[idx] - mean) * rsqrtf(var + EPSV) * gamma[n] + beta[n], 0.f);
  obf[(long long)b * ldo + ooff + n] = (bf16)y;
  if (of) of[idx] = y;
}

// ---------------------------------------------------------------------------
extern "C" void kernel_launch(void* const* d_in, const int* in_sizes, int n_in,
                              void* d_out, int out_size, void* d_ws, size_t ws_size,
                              hipStream_t stream)
{
  const float* last_pos = (const float*)d_in[0];
  const float* last_rel = (const float*)d_in[1];
  const float* h0   = (const float*)d_in[2];
  const float* c0   = (const float*)d_in[3];
  const float* W_se = (const float*)d_in[5];
  const float* b_se = (const float*)d_in[6];
  const float* W_hp = (const float*)d_in[7];
  const float* b_hp = (const float*)d_in[8];
  const float* Wih  = (const float*)d_in[9];
  const float* Whh  = (const float*)d_in[10];
  const float* bih  = (const float*)d_in[11];
  const float* bhh  = (const float*)d_in[12];
  const float* Wp_se = (const float*)d_in[13];
  const float* bp_se = (const float*)d_in[14];
  const float* Wp1  = (const float*)d_in[15];
  const float* bp1  = (const float*)d_in[16];
  const float* gp1  = (const float*)d_in[17];
  const float* betap1 = (const float*)d_in[18];
  const float* Wp2  = (const float*)d_in[19];
  const float* bp2  = (const float*)d_in[20];
  const float* gp2  = (const float*)d_in[21];
  const float* betap2 = (const float*)d_in[22];
  const float* Wm1  = (const float*)d_in[23];
  const float* bm1  = (const float*)d_in[24];
  const float* gm1  = (const float*)d_in[25];
  const float* betam1 = (const float*)d_in[26];
  const float* Wm2  = (const float*)d_in[27];
  const float* bm2  = (const float*)d_in[28];
  const float* gm2  = (const float*)d_in[29];
  const float* betam2 = (const float*)d_in[30];
  float* out = (float*)d_out;

  // ---- workspace carve-out (deterministic) ----
  char* wsp = (char*)d_ws;
  auto carve = [&](size_t bytes) -> char* {
    char* p = wsp;
    wsp += (bytes + 255) & ~(size_t)255;
    return p;
  };
  float* c_buf   = (float*)carve((size_t)BTOT * HD * 4);        // LSTM c
  bf16*  acat    = (bf16*)carve((size_t)BTOT * 192 * 2);        // [dec_in|h]
  bf16*  hcat    = (bf16*)carve((size_t)BTOT * 256 * 2);        // [h_new|pool_h]
  float* lastpos = (float*)carve((size_t)BTOT * 2 * 4);
  float* gates   = (float*)carve((size_t)BTOT * 512 * 4);
  float* hproj   = (float*)carve((size_t)BTOT * 512 * 4);
  bf16*  x1      = (bf16*)carve((size_t)GRP * 256 * 512 * 2);
  float* x2pre   = (float*)carve((size_t)GRP * 256 * 128 * 4);
  float* dh1pre  = (float*)carve((size_t)BTOT * 1024 * 4);
  bf16*  dh1bf   = (bf16*)carve((size_t)BTOT * 1024 * 2);
  float* dh2pre  = (float*)carve((size_t)BTOT * 128 * 4);
  float* dh_buf  = (float*)carve((size_t)BTOT * 128 * 4);
  float* stats1  = (float*)carve(2 * 1024 * 4);
  float* stats2  = (float*)carve(2 * 128 * 4);
  float* biasG   = (float*)carve(512 * 4);
  float* Wc      = (float*)carve(2 * 512 * 4);
  float* cbias   = (float*)carve(512 * 4);
  bf16*  WcatF   = (bf16*)carve((size_t)192 * 512 * 2);
  bf16*  Wp1hF   = (bf16*)carve((size_t)128 * 512 * 2);
  bf16*  Wp2F    = (bf16*)carve((size_t)512 * 128 * 2);
  bf16*  Wm1F    = (bf16*)carve((size_t)256 * 1024 * 2);
  bf16*  Wm2F    = (bf16*)carve((size_t)1024 * 128 * 2);
  (void)in_sizes; (void)n_in; (void)out_size; (void)ws_size;

  const int BIG = 1 << 30;

  // ---- one-time (per call) weight prep ----
  k_swizzle<<<384, 256, 0, stream>>>(Wih, Whh, 64, 512, 6, WcatF, 192 * 512);
  k_swizzle<<<256, 256, 0, stream>>>(Wp1 + 64 * 512, Wp1, BIG, 512, 4, Wp1hF, 128 * 512);
  k_swizzle<<<256, 256, 0, stream>>>(Wp2, Wp2, BIG, 128, 16, Wp2F, 512 * 128);
  k_swizzle<<<1024, 256, 0, stream>>>(Wm1, Wm1, BIG, 1024, 8, Wm1F, 256 * 1024);
  k_swizzle<<<512, 256, 0, stream>>>(Wm2, Wm2, BIG, 128, 32, Wm2F, 1024 * 128);
  k_biascomb<<<2, 256, 0, stream>>>(bih, bhh, biasG);
  k_wc<<<2, 256, 0, stream>>>(Wp_se, bp_se, Wp1, bp1, Wc, cbias);
  k_init<<<BTOT, 128, 0, stream>>>(h0, c0, last_pos, last_rel, W_se, b_se,
                                   c_buf, acat, lastpos);

  for (int t = 0; t < T_STEPS; ++t) {
    // LSTM gates: [dec_in|h](B,192) @ [Wih;Whh](192,512)  -> gates
    // waves = Mtiles(256) * Ntiles/2(16) = 4096 -> 512 blocks
    k_gemm<<<dim3(512, 1), 256, 0, stream>>>(acat, 192, 0, WcatF, biasG,
                                             gates, 512, 0, 32, 6);
    // cell + hidden2pos + pos update + spatial embedding (writes traj[t])
    k_cell<<<BTOT, 128, 0, stream>>>(gates, c_buf, hcat, acat, lastpos,
                                     out + (size_t)t * BTOT * 2,
                                     W_hp, b_hp, W_se, b_se);
    // hproj = h_new @ Wp1[64:] + (bp_se@Wp1[:64] + bp1)
    k_gemm<<<dim3(512, 1), 256, 0, stream>>>(hcat, 256, 0, Wp1hF, cbias,
                                             hproj, 512, 0, 32, 4);
    // pool layer 1: + rel@Wc, per-group BN+ReLU -> bf16 x1
    k_pool1<<<dim3(2, GRP), 256, 0, stream>>>(hproj, lastpos, Wc, gp1, betap1, x1);
    // pool layer 2 GEMM (grouped): x1(256,512) @ Wp2(512,128) + bp2
    // waves/group = 16 * 4 = 64 -> 8 blocks per group
    k_gemm<<<dim3(8, GRP), 256, 0, stream>>>(x1, 512, 256LL * 512, Wp2F, bp2,
                                             x2pre, 128, 256LL * 128, 8, 16);
    // per-group BN + ReLU + max over peds -> hcat[:,128:256]
    k_pool_bnmax<<<GRP, 128, 0, stream>>>(x2pre, gp2, betap2, hcat);
    // merge layer 1: [h_new|pool_h](B,256) @ Wm1(256,1024) + bm1
    // waves = 256 * 32 = 8192 -> 1024 blocks
    k_gemm<<<dim3(1024, 1), 256, 0, stream>>>(hcat, 256, 0, Wm1F, bm1,
                                              dh1pre, 1024, 0, 64, 8);
    k_zero<<<8, 256, 0, stream>>>(stats1, 2048);
    k_colstats<<<dim3(16, 4), 256, 0, stream>>>(dh1pre, 1024, stats1);
    k_bnrelu<<<16384, 256, 0, stream>>>(dh1pre, 1024, stats1, gm1, betam1,
                                        dh1bf, 1024, 0, nullptr);
    // merge layer 2: dh1(B,1024) @ Wm2(1024,128) + bm2
    // waves = 256 * 4 = 1024 -> 128 blocks
    k_gemm<<<dim3(128, 1), 256, 0, stream>>>(dh1bf, 1024, 0, Wm2F, bm2,
                                             dh2pre, 128, 0, 8, 32);
    k_zero<<<1, 256, 0, stream>>>(stats2, 256);
    k_colstats<<<dim3(2, 4), 256, 0, stream>>>(dh2pre, 128, stats2);
    // dh -> fp32 carry + bf16 into acat h-slot for next LSTM step
    k_bnrelu<<<2048, 256, 0, stream>>>(dh2pre, 128, stats2, gm2, betam2,
                                       acat, 192, 64, dh_buf);
  }

  // final hidden state (dh after last step) -> out[T*B*2 ...]
  k_copy<<<2048, 256, 0, stream>>>(dh_buf, out + (size_t)T_STEPS * BTOT * 2,
                                   BTOT * HD);
}
